// RNN_27797028340053
// MI455X (gfx1250) — compile-verified, matrix operand order checked by reading, not run on hardware
//
#include <hip/hip_runtime.h>
#include <hip/hip_bf16.h>
#include <stdint.h>

// ---------------------------------------------------------------------------
// RNN (4-layer recurrent, DIM=1024, B=4, L=1024, VOCAB=32000) for MI455X.
//
// Phase 1 (recurrence): anti-diagonal wavefront over (t, layer); 64 blocks x
//   256 threads = 512 wave32's, wave (layer,nt) owns one 16x16 output tile
//   with full-K (2048) accumulation via v_wmma_f32_16x16x32_bf16.
//   Weights pre-swizzled to WMMA B-fragment layout (32B/lane contiguous),
//   L2-resident (32MB bf16 << 192MB L2). A staged in LDS per diagonal.
//   K-loop is software-pipelined (double-buffered A/B fragments) so the
//   pre-WMMA wait covers only the in-flight next-iteration loads.
//   1027 device-wide syncs (atomic counter + generation, s_sleep spin).
// Phase 2 (projection): [4096,1024]x[1024,32000] bf16 WMMA GEMM, 128x128
//   C-block per workgroup, fp32 logits out (524MB write dominates).
// ---------------------------------------------------------------------------

typedef unsigned short ushort_t;
typedef __attribute__((ext_vector_type(16))) __bf16   v16bf;
typedef __attribute__((ext_vector_type(8)))  float    v8f;
typedef __attribute__((ext_vector_type(8)))  uint32_t v8u;
typedef __attribute__((ext_vector_type(4)))  uint32_t v4u;

#define DIMD    1024
#define TWO_D   2048
#define NLAYER  4
#define NBATCH  4
#define SEQL    1024
#define NVOCAB  32000
#define NT_REC  128   // 2048/16 N tiles per layer
#define KC_REC  64    // 2048/32 K chunks
#define NT_PRJ  2000  // 32000/16
#define KC_PRJ  32    // 1024/32
#define RBLOCKS 64
#define RTHREADS 256

// ws layout (bytes)
#define OFF_SYNC 0u
#define OFF_S    ((size_t)4096)                       // s_buf bf16 [4][2][16][1024] = 256KB
#define OFF_H    (OFF_S  + (size_t)262144)            // h_buf bf16 [4][2][16][1024] = 256KB
#define OFF_X    (OFF_H  + (size_t)262144)            // x    bf16 [1024][16][1024]  = 32MB
#define OFF_YS   (OFF_X  + (size_t)33554432)          // ys   bf16 [1024][16][1024]  = 32MB
#define OFF_WSW  (OFF_YS + (size_t)33554432)          // Wsw  bf16 [4][128][64][32][16] = 32MB
#define OFF_PSW  (OFF_WSW+ (size_t)33554432)          // Psw  bf16 [2000][32][32][16]   = 62.5MB

// K index of element e (0..15) of a 16-bit A/B WMMA fragment, for lane half.
__device__ __forceinline__ int kmap(int e, int half) {
  int v = e >> 1, pos = e & 1;
  int k = (v < 4) ? (v * 2 + pos) : (16 + (v - 4) * 2 + pos);
  return k + half * 8;
}

__device__ __forceinline__ ushort_t f2bf(float f) {
  union { float f; uint32_t u; } x; x.f = f;
  uint32_t u = x.u;
  u += 0x7fffu + ((u >> 16) & 1u);   // round-to-nearest-even
  return (ushort_t)(u >> 16);
}

// ---------------------------------------------------------------------------
// init: zero sync words, fill s_buf (both parities) from starter (rows>=4 = 0)
__global__ void k_init(const float* __restrict__ starter,
                       ushort_t* __restrict__ sbuf,
                       unsigned* __restrict__ sync) {
  int i = blockIdx.x * blockDim.x + threadIdx.x;
  if (i < 64) sync[i] = 0u;
  if (i < NLAYER * 2 * 16 * DIMD) {
    int k = i & (DIMD - 1);
    int m = (i >> 10) & 15;
    int layer = i >> 15;
    float v = (m < NBATCH) ? starter[layer * DIMD + k] : 0.f;
    sbuf[i] = f2bf(v);
  }
}

// embedding gather -> bf16 x[l][16][1024], rows 4..15 zero
__global__ void k_embed(const int* __restrict__ ids,
                        const float* __restrict__ embed,
                        ushort_t* __restrict__ xb) {
  size_t i = (size_t)blockIdx.x * blockDim.x + threadIdx.x;
  if (i >= (size_t)SEQL * 16 * DIMD) return;
  int k = (int)(i & (DIMD - 1));
  int m = (int)((i >> 10) & 15);
  int l = (int)(i >> 14);
  float v = 0.f;
  if (m < NBATCH) v = embed[(size_t)ids[m * SEQL + l] * DIMD + k];
  xb[i] = f2bf(v);
}

// W [4][2048][2048] f32 -> bf16 swizzled B-fragments [layer][nt][kc][lane][16]
__global__ void k_wswz(const float* __restrict__ W, ushort_t* __restrict__ wsw) {
  int i = blockIdx.x * blockDim.x + threadIdx.x;            // (layer,nt,kc,lane)
  if (i >= NLAYER * NT_REC * KC_REC * 32) return;
  int lane = i & 31, kc = (i >> 5) & 63, nt = (i >> 11) & 127, layer = i >> 18;
  int half = lane >> 4;
  int n = nt * 16 + (lane & 15);
  const float* wrow = W + ((size_t)layer * TWO_D + n) * TWO_D; // Wi row n
  ushort_t* dst = wsw + (size_t)i * 16;
#pragma unroll
  for (int e = 0; e < 16; ++e)
    dst[e] = f2bf(wrow[kc * 32 + kmap(e, half)]);             // B[k][n] = Wi[n][k]
}

// out_w [32000][1024] f32 -> bf16 swizzled B-fragments [nt][kc][lane][16]
__global__ void k_pswz(const float* __restrict__ OW, ushort_t* __restrict__ psw) {
  int i = blockIdx.x * blockDim.x + threadIdx.x;             // (nt,kc,lane)
  if (i >= NT_PRJ * KC_PRJ * 32) return;
  int lane = i & 31, kc = (i >> 5) & 31, nt = i >> 10;
  int half = lane >> 4;
  int v = nt * 16 + (lane & 15);
  const float* wrow = OW + (size_t)v * DIMD;
  ushort_t* dst = psw + (size_t)i * 16;
#pragma unroll
  for (int e = 0; e < 16; ++e)
    dst[e] = f2bf(wrow[kc * 32 + kmap(e, half)]);
}

// ---------------------------------------------------------------------------
// Persistent recurrence over anti-diagonals d = t + layer.
__global__ void __launch_bounds__(RTHREADS)
rnn_recurrent(const ushort_t* __restrict__ xbuf, ushort_t* __restrict__ ys,
              ushort_t* __restrict__ hbuf, ushort_t* __restrict__ sbuf,
              const ushort_t* __restrict__ wsw, unsigned* __restrict__ sync) {
  extern __shared__ ushort_t Ash[];            // A = concat(h,s): [16][2048] bf16 = 64KB
  const int tid   = threadIdx.x;
  const int wave  = tid >> 5;
  const int lane  = tid & 31;
  const int layer = blockIdx.x >> 4;           // 16 blocks per layer
  const int nt    = (blockIdx.x & 15) * 8 + wave;
  const int half  = lane >> 4;
  const int ncol  = lane & 15;                 // A row (M) on load side, D col (N) on store side
  const ushort_t* wl = wsw + (size_t)(layer * NT_REC + nt) * KC_REC * 512;
  unsigned* counter = sync;
  unsigned* gen     = sync + 16;

  for (int d = 0; d < SEQL + NLAYER - 1; ++d) {
    const int t = d - layer;
    const bool active = (t >= 0) && (t < SEQL);

    if (active) {  // stage A into LDS: rows m, cols 0..1023=h_in, 1024..2047=s_in
      const ushort_t* hin = (layer == 0)
          ? (xbuf + (size_t)t * (16 * DIMD))
          : (hbuf + ((size_t)(layer - 1) * 2 + (t & 1)) * (16 * DIMD));
      const ushort_t* sin = sbuf + ((size_t)layer * 2 + (t & 1)) * (16 * DIMD);
      const v4u* h4 = (const v4u*)hin;
      const v4u* s4 = (const v4u*)sin;
      v4u* a4 = (v4u*)Ash;
      for (int c = tid; c < 4096; c += RTHREADS) {     // 4096 x 16B = 64KB
        int m = c >> 8, q = c & 255;                   // 256 uint4 per row
        a4[c] = (q < 128) ? h4[m * 128 + q] : s4[m * 128 + (q - 128)];
      }
    }
    __syncthreads();

    if (active) {
      v8f acc = {0.f, 0.f, 0.f, 0.f, 0.f, 0.f, 0.f, 0.f};
      const uint32_t* arow = (const uint32_t*)(Ash + ncol * TWO_D);

      // software-pipelined K loop: loads for kc+1 issued before WMMA on kc
      v8u au, bu;
#pragma unroll
      for (int v = 0; v < 8; ++v)
        au[v] = arow[kmap(v * 2, half) >> 1];
      bu = *(const v8u*)(wl + (size_t)lane * 16);

      for (int kc = 0; kc < KC_REC; ++kc) {
        const int kn = (kc + 1 < KC_REC) ? kc + 1 : kc;       // clamped (branchless)
        v8u aun, bun;
#pragma unroll
        for (int v = 0; v < 8; ++v)
          aun[v] = arow[(kn * 32 + kmap(v * 2, half)) >> 1];
        bun = *(const v8u*)(wl + ((size_t)kn * 32 + lane) * 16);
        const int kp = (kc + 8 < KC_REC) ? kc + 8 : KC_REC - 1;
        __builtin_prefetch((const void*)(wl + ((size_t)kp * 32 + lane) * 16), 0, 1);

        v16bf a = __builtin_bit_cast(v16bf, au);
        v16bf b = __builtin_bit_cast(v16bf, bu);
        acc = __builtin_amdgcn_wmma_f32_16x16x32_bf16(
            false, a, false, b, (short)0, acc, false, false);
        au = aun; bu = bun;
      }

      // scatter D tile (bf16): nt<64 -> h part, nt>=64 -> s part
      ushort_t* dst;
      int col;
      if (nt < 64) {
        col = nt * 16 + ncol;
        dst = (layer == NLAYER - 1)
            ? (ys + (size_t)t * (16 * DIMD))
            : (hbuf + ((size_t)layer * 2 + (t & 1)) * (16 * DIMD));
      } else {
        col = (nt - 64) * 16 + ncol;
        dst = sbuf + ((size_t)layer * 2 + ((t + 1) & 1)) * (16 * DIMD);
      }
#pragma unroll
      for (int r = 0; r < 8; ++r)
        dst[(r + half * 8) * DIMD + col] = f2bf(acc[r]);
    }

    // device-wide release/acquire sync (monotonic counter, no reset needed)
    __threadfence();
    __syncthreads();
    if (tid == 0) {
      const unsigned target = (unsigned)(d + 1);
      if (atomicAdd(counter, 1u) == (unsigned)RBLOCKS * target - 1u) {
        __hip_atomic_store(gen, target, __ATOMIC_RELEASE, __HIP_MEMORY_SCOPE_AGENT);
      } else {
        while (__hip_atomic_load(gen, __ATOMIC_ACQUIRE, __HIP_MEMORY_SCOPE_AGENT) < target)
          __builtin_amdgcn_s_sleep(2);
      }
    }
    __syncthreads();
  }
}

// ---------------------------------------------------------------------------
// Projection: logits[b][l][v] = sum_k ys[l][b][k] * out_w[v][k]  (M=4096 rows)
__global__ void __launch_bounds__(256)
rnn_proj(const ushort_t* __restrict__ ysb, const ushort_t* __restrict__ psw,
         float* __restrict__ out) {
  const int wave  = threadIdx.x >> 5;
  const int lane  = threadIdx.x & 31;
  const int half  = lane >> 4;
  const int l15   = lane & 15;
  const int mtile = blockIdx.x * 8 + wave;   // 0..255
  const int m     = mtile * 16 + l15;        // A row for loads
  const uint32_t* arow =
      (const uint32_t*)(ysb + ((size_t)(m >> 2) * 16 + (m & 3)) * DIMD);
  const ushort_t* pbase = psw + (((size_t)blockIdx.y * 8) * KC_PRJ + 0) * 512;

  v8f acc[8];
#pragma unroll
  for (int j = 0; j < 8; ++j) acc[j] = (v8f){0.f, 0.f, 0.f, 0.f, 0.f, 0.f, 0.f, 0.f};

  // A double-buffered across kc; all 8 B fragments of a K-step loaded as a
  // batch before the 8 WMMAs so the first WMMA waits at loadcnt<=7, not 0.
  v8u au;
#pragma unroll
  for (int v = 0; v < 8; ++v)
    au[v] = arow[kmap(v * 2, half) >> 1];

  for (int kc = 0; kc < KC_PRJ; ++kc) {
    v8u bu[8];
#pragma unroll
    for (int j = 0; j < 8; ++j)
      bu[j] = *(const v8u*)(pbase + ((size_t)j * KC_PRJ + kc) * 512 + (size_t)lane * 16);

    const int kn = (kc + 1 < KC_PRJ) ? kc + 1 : kc;
    v8u aun;
#pragma unroll
    for (int v = 0; v < 8; ++v)
      aun[v] = arow[(kn * 32 + kmap(v * 2, half)) >> 1];

    v16bf a = __builtin_bit_cast(v16bf, au);
#pragma unroll
    for (int j = 0; j < 8; ++j) {
      v16bf b = __builtin_bit_cast(v16bf, bu[j]);
      acc[j] = __builtin_amdgcn_wmma_f32_16x16x32_bf16(
          false, a, false, b, (short)0, acc[j], false, false);
    }
    au = aun;
  }
#pragma unroll
  for (int j = 0; j < 8; ++j) {
    int vcol = (blockIdx.y * 8 + j) * 16 + l15;
#pragma unroll
    for (int r = 0; r < 8; ++r) {
      int mr = mtile * 16 + r + half * 8;
      int l = mr >> 2, b = mr & 3;
      out[((size_t)b * SEQL + l) * NVOCAB + vcol] = acc[j][r];
    }
  }
}

// ---------------------------------------------------------------------------
extern "C" void kernel_launch(void* const* d_in, const int* in_sizes, int n_in,
                              void* d_out, int out_size, void* d_ws, size_t ws_size,
                              hipStream_t stream) {
  const int*   ids     = (const int*)d_in[0];
  const float* embed   = (const float*)d_in[1];
  const float* W       = (const float*)d_in[2];
  const float* out_w   = (const float*)d_in[3];
  const float* starter = (const float*)d_in[4];
  float* out = (float*)d_out;
  char*  ws  = (char*)d_ws;

  unsigned* sync = (unsigned*)(ws + OFF_SYNC);
  ushort_t* sbuf = (ushort_t*)(ws + OFF_S);
  ushort_t* hbuf = (ushort_t*)(ws + OFF_H);
  ushort_t* xbuf = (ushort_t*)(ws + OFF_X);
  ushort_t* ysb  = (ushort_t*)(ws + OFF_YS);
  ushort_t* wsw  = (ushort_t*)(ws + OFF_WSW);
  ushort_t* psw  = (ushort_t*)(ws + OFF_PSW);

  k_init<<<512, 256, 0, stream>>>(starter, sbuf, sync);
  k_embed<<<(SEQL * 16 * DIMD) / 256, 256, 0, stream>>>(ids, embed, xbuf);
  k_wswz<<<(NLAYER * NT_REC * KC_REC * 32) / 256, 256, 0, stream>>>(W, wsw);
  k_pswz<<<(NT_PRJ * KC_PRJ * 32) / 256, 256, 0, stream>>>(out_w, psw);
  rnn_recurrent<<<RBLOCKS, RTHREADS, 16 * TWO_D * 2, stream>>>(xbuf, ysb, hbuf, sbuf, wsw, sync);
  rnn_proj<<<dim3(32, 250), 256, 0, stream>>>(ysb, psw, out);
}
